// GraphAttentionLayer_9431748182561
// MI455X (gfx1250) — matrix operand order repaired and checked by reading, not compiled, not run on hardware
//
#include <hip/hip_runtime.h>
#include <hip/hip_bf16.h>

// ---------------------------------------------------------------------------
// GAT layer for MI455X (gfx1250, wave32, WMMA).
// Strategy: never materialize the 8192x8192 attention matrix. Stream the
// 268MB support matrix twice (softmax-stats pass + fused attention GEMM pass),
// keep h resident as a 3.4MB bf16 transposed copy (L2-resident), and do the
// 27 GFLOP attention@h GEMM with v_wmma_f32_16x16x32_bf16 building the bf16
// attention A-tiles on the fly.
// ---------------------------------------------------------------------------

typedef __attribute__((ext_vector_type(16))) __bf16 v16bf;
typedef __attribute__((ext_vector_type(8)))  __bf16 v8bf;
typedef __attribute__((ext_vector_type(8)))  float  v8f;

#define N_ROWS 8192
#define IN_F   300
#define IN_FP  320      // padded K for gemm1 (multiple of 32)
#define OUT_F  200
#define OUT_FP 208      // padded N (13 tiles of 16)
#define N_TILES 13
#define ALPHA  0.2f
#define NEG_BIG (-9000000000000000.0f)

// ---- workspace layout (bytes, 256B-aligned) ----
#define OFF_XPAD 0u                       // 8192*320 bf16   = 5,242,880
#define OFF_WT   5242880u                 // 208*320 bf16    =   133,120
#define OFF_H    5376000u                 // 8192*200 f32    = 6,553,600
#define OFF_HT   11929600u                // 208*8192 bf16   = 3,407,872
#define OFF_S1   15337472u                // 8192 f32
#define OFF_S2   15370240u
#define OFF_RMAX 15403008u
#define OFF_RSUM 15435776u

static __device__ __forceinline__ v8f wmma_bf16(v16bf a, v16bf b, v8f c) {
    return __builtin_amdgcn_wmma_f32_16x16x32_bf16(
        /*neg_a=*/false, a, /*neg_b=*/false, b,
        /*c_mod=*/(short)0, c, /*reuse_a=*/false, /*reuse_b=*/false);
}

// ---------------------------------------------------------------------------
// prep: pad x -> bf16 [8192 x 320] row-major (A operand feed for gemm1)
// ---------------------------------------------------------------------------
__global__ void gat_prep_x(const float* __restrict__ x, __bf16* __restrict__ xpad) {
    int idx = blockIdx.x * 256 + threadIdx.x;
    if (idx >= N_ROWS * IN_FP) return;
    int i = idx / IN_FP, k = idx - i * IN_FP;
    float v = (k < IN_F) ? x[(size_t)i * IN_F + k] : 0.0f;
    xpad[idx] = (__bf16)v;
}

// prep: W (300x200) -> WT bf16 [208 x 320] row-major (B operand feed, K contiguous)
__global__ void gat_prep_wt(const float* __restrict__ W, __bf16* __restrict__ wt) {
    int idx = blockIdx.x * 256 + threadIdx.x;
    if (idx >= OUT_FP * IN_FP) return;
    int f = idx / IN_FP, k = idx - f * IN_FP;
    float v = (f < OUT_F && k < IN_F) ? W[(size_t)k * OUT_F + f] : 0.0f;
    wt[idx] = (__bf16)v;
}

// ---------------------------------------------------------------------------
// gemm1: h = x @ W via WMMA bf16. One wave per (16-row block, 16-col tile).
// Writes h fp32 [8192x200] and hT bf16 [208x8192] (transposed, K-contiguous
// B-operand layout for the attention GEMM; rows 200..207 become zero since
// WT rows >=200 are zero).
// ---------------------------------------------------------------------------
__global__ __launch_bounds__(256) void gat_gemm1(
    const __bf16* __restrict__ xpad, const __bf16* __restrict__ wt,
    float* __restrict__ h, __bf16* __restrict__ hT)
{
    const int lane = threadIdx.x & 31;
    const int wg   = blockIdx.x * 8 + (threadIdx.x >> 5);   // 0..6655
    const int rb   = wg & 511;       // row block 0..511
    const int nt   = wg >> 9;        // col tile  0..12
    const int M    = lane & 15;
    const int half = lane >> 4;
    const int i0   = rb * 16;

    v8f zero;
#pragma unroll
    for (int r = 0; r < 8; ++r) zero[r] = 0.0f;
    v8f c = zero;

    for (int k0 = 0; k0 < IN_FP; k0 += 32) {
        // A: lane holds row M, K = {0..15} (lanes<16) / {16..31}
        const __bf16* ap = xpad + (size_t)(i0 + M) * IN_FP + k0 + half * 16;
        v8bf a0 = *(const v8bf*)(ap);
        v8bf a1 = *(const v8bf*)(ap + 8);
        v16bf A;
#pragma unroll
        for (int i = 0; i < 8; ++i) { A[i] = a0[i]; A[i + 8] = a1[i]; }
        // B: lane holds col N = lane&15, K = 16 contiguous (lo/hi half)
        const __bf16* bp = wt + (size_t)(nt * 16 + M) * IN_FP + k0 + half * 16;
        v16bf B = *(const v16bf*)bp;
        c = wmma_bf16(A, B, c);
    }

#pragma unroll
    for (int r = 0; r < 8; ++r) {
        int i   = i0 + r + half * 8;       // C layout: VGPR r -> row r (+8 hi half)
        int col = nt * 16 + M;             //           lane%16 -> col
        hT[(size_t)col * N_ROWS + i] = (__bf16)c[r];
        if (col < OUT_F) h[(size_t)i * OUT_F + col] = c[r];
    }
}

// ---------------------------------------------------------------------------
// s1[i] = h[i,:] . a[:200],  s2[i] = h[i,:] . a[200:400].  Wave per row.
// ---------------------------------------------------------------------------
__global__ __launch_bounds__(256) void gat_s12(
    const float* __restrict__ h, const float* __restrict__ a,
    float* __restrict__ s1, float* __restrict__ s2)
{
    const int lane = threadIdx.x & 31;
    const int row  = blockIdx.x * 8 + (threadIdx.x >> 5);
    const float* hr = h + (size_t)row * OUT_F;
    float p1 = 0.0f, p2 = 0.0f;
    for (int f = lane; f < OUT_F; f += 32) {
        float hv = hr[f];
        p1 += hv * a[f];
        p2 += hv * a[OUT_F + f];
    }
#pragma unroll
    for (int off = 16; off; off >>= 1) {
        p1 += __shfl_xor(p1, off, 32);
        p2 += __shfl_xor(p2, off, 32);
    }
    if (lane == 0) { s1[row] = p1; s2[row] = p2; }
}

// ---------------------------------------------------------------------------
// Row softmax stats: online max + rescaled sum over masked leaky-relu scores.
// Wave per row; fully coalesced streaming of the support row.
// ---------------------------------------------------------------------------
__global__ __launch_bounds__(256) void gat_stats(
    const int* __restrict__ support, const float* __restrict__ s1,
    const float* __restrict__ s2, float* __restrict__ rmax, float* __restrict__ rsum)
{
    const int lane = threadIdx.x & 31;
    const int row  = blockIdx.x * 8 + (threadIdx.x >> 5);
    const int* sr  = support + (size_t)row * N_ROWS;
    const float s1v = s1[row];

    float m = -INFINITY, l = 0.0f;
    for (int j = lane; j < N_ROWS; j += 32) {
        float t = s1v + s2[j];
        t = fmaxf(t, ALPHA * t);                 // leaky relu
        float v = (sr[j] > 0) ? t : NEG_BIG;     // mask
        float nm = fmaxf(m, v);
        l = l * __expf(m - nm) + __expf(v - nm);
        m = nm;
    }
#pragma unroll
    for (int off = 16; off; off >>= 1) {
        float mo = __shfl_xor(m, off, 32);
        float lo = __shfl_xor(l, off, 32);
        float nm = fmaxf(m, mo);
        l = l * __expf(m - nm) + lo * __expf(mo - nm);
        m = nm;
    }
    if (lane == 0) { rmax[row] = m; rsum[row] = l; }
}

static __device__ __forceinline__ __bf16 attn_val(int sup, float s2v, float s1r, float mr) {
    float t = s1r + s2v;
    t = fmaxf(t, ALPHA * t);
    float p = (sup > 0) ? __expf(t - mr) : 0.0f;   // exp(score - rowmax), masked -> 0
    return (__bf16)p;
}

// ---------------------------------------------------------------------------
// Fused attention GEMM: out = elu( softmax(masked) @ h ).
// One 16-row block per workgroup; 4 waves split K=8192 (2048 each); each wave
// holds all 13 N-tiles (16x16 f32 accumulators) so support is read once.
// A-tiles (bf16 attention probs) are built on the fly in contiguous K order
// (lane half h: K 16h+s); B-tiles come from hT (K-contiguous v16bf loads).
// ---------------------------------------------------------------------------
__global__ __launch_bounds__(128) void gat_attn_gemm(
    const int* __restrict__ support, const float* __restrict__ s1,
    const float* __restrict__ s2, const float* __restrict__ rmax,
    const float* __restrict__ rsum, const __bf16* __restrict__ hT,
    float* __restrict__ out)
{
    __shared__ float red[3 * 256];
    const int lane = threadIdx.x & 31;
    const int w    = threadIdx.x >> 5;           // wave 0..3 -> K split
    const int i0   = blockIdx.x * 16;
    const int M    = lane & 15;
    const int half = lane >> 4;
    const int row  = i0 + M;

    const float s1r = s1[row];
    const float mr  = rmax[row];
    const int* srow = support + (size_t)row * N_ROWS;

    v8f zero;
#pragma unroll
    for (int r = 0; r < 8; ++r) zero[r] = 0.0f;
    v8f c[N_TILES];
#pragma unroll
    for (int t = 0; t < N_TILES; ++t) c[t] = zero;

    const int kbase = w * 2048;
    for (int ks = 0; ks < 64; ++ks) {
        const int k0   = kbase + ks * 32;
        const int seg0 = k0 + half * 16;         // A layout: lanes<16 get K 0..15
        const int seg1 = seg0 + 8;               //           lanes>=16 get K 16..31

        int4 sa = *(const int4*)(srow + seg0);
        int4 sb = *(const int4*)(srow + seg0 + 4);
        int4 sc = *(const int4*)(srow + seg1);
        int4 sd = *(const int4*)(srow + seg1 + 4);
        float4 za = *(const float4*)(s2 + seg0);
        float4 zb = *(const float4*)(s2 + seg0 + 4);
        float4 zc = *(const float4*)(s2 + seg1);
        float4 zd = *(const float4*)(s2 + seg1 + 4);

        v16bf A;
        A[0]  = attn_val(sa.x, za.x, s1r, mr);
        A[1]  = attn_val(sa.y, za.y, s1r, mr);
        A[2]  = attn_val(sa.z, za.z, s1r, mr);
        A[3]  = attn_val(sa.w, za.w, s1r, mr);
        A[4]  = attn_val(sb.x, zb.x, s1r, mr);
        A[5]  = attn_val(sb.y, zb.y, s1r, mr);
        A[6]  = attn_val(sb.z, zb.z, s1r, mr);
        A[7]  = attn_val(sb.w, zb.w, s1r, mr);
        A[8]  = attn_val(sc.x, zc.x, s1r, mr);
        A[9]  = attn_val(sc.y, zc.y, s1r, mr);
        A[10] = attn_val(sc.z, zc.z, s1r, mr);
        A[11] = attn_val(sc.w, zc.w, s1r, mr);
        A[12] = attn_val(sd.x, zd.x, s1r, mr);
        A[13] = attn_val(sd.y, zd.y, s1r, mr);
        A[14] = attn_val(sd.z, zd.z, s1r, mr);
        A[15] = attn_val(sd.w, zd.w, s1r, mr);

#pragma unroll
        for (int t = 0; t < N_TILES; ++t) {
            const __bf16* bp = hT + (size_t)(t * 16 + M) * N_ROWS + k0 + half * 16;
            v16bf B = *(const v16bf*)bp;         // 32B aligned, 2x b128
            c[t] = wmma_bf16(A, B, c[t]);
        }
    }

    // cross-wave K reduction in LDS, then normalize (1/rowsum) + elu + store
#pragma unroll
    for (int t = 0; t < N_TILES; ++t) {
        if (w > 0) {
#pragma unroll
            for (int r = 0; r < 8; ++r)
                red[(w - 1) * 256 + lane * 8 + r] = c[t][r];
        }
        __syncthreads();
        if (w == 0) {
#pragma unroll
            for (int r = 0; r < 8; ++r) {
                float v = c[t][r]
                        + red[        lane * 8 + r]
                        + red[256   + lane * 8 + r]
                        + red[512   + lane * 8 + r];
                int i   = i0 + r + half * 8;
                int col = t * 16 + M;
                if (col < OUT_F) {
                    float hp = v * (1.0f / rsum[i]);
                    out[(size_t)i * OUT_F + col] = (hp > 0.0f) ? hp : (__expf(hp) - 1.0f);
                }
            }
        }
        __syncthreads();
    }
}

// ---------------------------------------------------------------------------
extern "C" void kernel_launch(void* const* d_in, const int* in_sizes, int n_in,
                              void* d_out, int out_size, void* d_ws, size_t ws_size,
                              hipStream_t stream) {
    const float* x       = (const float*)d_in[0];   // 8192*300
    const float* W       = (const float*)d_in[1];   // 300*200
    const float* a       = (const float*)d_in[2];   // 400
    const int*   support = (const int*)d_in[3];     // 8192*8192
    float* out = (float*)d_out;                     // 8192*200 f32

    char* ws = (char*)d_ws;
    __bf16* xpad = (__bf16*)(ws + OFF_XPAD);
    __bf16* wt   = (__bf16*)(ws + OFF_WT);
    float*  h    = (float*)(ws + OFF_H);
    __bf16* hT   = (__bf16*)(ws + OFF_HT);
    float*  s1   = (float*)(ws + OFF_S1);
    float*  s2   = (float*)(ws + OFF_S2);
    float*  rmax = (float*)(ws + OFF_RMAX);
    float*  rsum = (float*)(ws + OFF_RSUM);

    gat_prep_x <<<(N_ROWS * IN_FP + 255) / 256, 256, 0, stream>>>(x, xpad);
    gat_prep_wt<<<(OUT_FP * IN_FP + 255) / 256, 256, 0, stream>>>(W, wt);
    gat_gemm1  <<<(512 * N_TILES) / 8, 256, 0, stream>>>(xpad, wt, h, hT);
    gat_s12    <<<N_ROWS / 8, 256, 0, stream>>>(h, a, s1, s2);
    gat_stats  <<<N_ROWS / 8, 256, 0, stream>>>(support, s1, s2, rmax, rsum);
    gat_attn_gemm<<<N_ROWS / 16, 128, 0, stream>>>(support, s1, s2, rmax, rsum, hT, out);
}
